// PFNAttention_28948079575071
// MI455X (gfx1250) — compile-verified, matrix operand order checked by reading, not run on hardware
//
#include <hip/hip_runtime.h>
#include <math.h>

#define N_HEADS 8
#define D_MODEL 256
#define D_HEAD  32
#define BATCH   2
#define SEQ     2048

typedef __attribute__((ext_vector_type(16))) __bf16 v16bf;
typedef __attribute__((ext_vector_type(8)))  float  v8f;

// ---- f32 -> bf16 via hardware conversion ----
__device__ __forceinline__ unsigned short f2bf(float f) {
  union { __bf16 h; unsigned short u; } cv;
  cv.h = (__bf16)f;
  return cv.u;
}

// ---- WMMA fragment loader (CDNA5 wave32, 16x16x32 bf16) ----
// Works for A (rows) and for B when the B matrix is stored transposed
// (lane's 16 columns become 16 contiguous rows). Per lane: two 16-byte loads.
// lane L: row m = L&15 ; lanes 0-15 take K 0..7 & 16..23, lanes 16-31 take
// K 8..15 & 24..31.
__device__ __forceinline__ v16bf frag_rows16(const unsigned short* __restrict__ src,
                                             int ld) {
  const int m  = threadIdx.x & 15;
  const int kb = ((threadIdx.x >> 4) & 1) * 8;
  union { v16bf v; uint4 q[2]; } f;
  const unsigned short* p = src + (size_t)m * ld + kb;
  f.q[0] = *(const uint4*)(p);
  f.q[1] = *(const uint4*)(p + 16);
  return f.v;
}

__device__ __forceinline__ v8f wmma_bf16(v16bf a, v16bf b, v8f c) {
  return __builtin_amdgcn_wmma_f32_16x16x32_bf16(
      false, a, false, b, (short)0, c, false, false);
}

// ============ Kernel 0a: convert x -> bf16 (vectorized, 8 elems/thread) ====
__global__ void __launch_bounds__(256) cvt_x_kernel(
    const float* __restrict__ x, unsigned short* __restrict__ Xb, int n8) {
  int t = blockIdx.x * blockDim.x + threadIdx.x;
  if (t >= n8) return;
  const float4* src = (const float4*)x + (size_t)t * 2;
  float4 a = src[0];
  float4 b = src[1];
  union { unsigned short us[8]; uint4 q; } o;
  o.us[0] = f2bf(a.x); o.us[1] = f2bf(a.y); o.us[2] = f2bf(a.z); o.us[3] = f2bf(a.w);
  o.us[4] = f2bf(b.x); o.us[5] = f2bf(b.y); o.us[6] = f2bf(b.z); o.us[7] = f2bf(b.w);
  ((uint4*)Xb)[t] = o.q;
}

// ============ Kernel 0b: convert + transpose weights to bf16 ==============
// Output layout (u16): WqT|WkT|WvT each [H*32 rows][256] (row = h*32+n),
// then WoT [256 rows][256] (row n, col k = W_O[k*256+n]).
__global__ void __launch_bounds__(256) cvt_w_kernel(
    const float* __restrict__ Wq, const float* __restrict__ Wk,
    const float* __restrict__ Wv, const float* __restrict__ Wo,
    unsigned short* __restrict__ Wt) {
  int t = blockIdx.x * blockDim.x + threadIdx.x;
  const int perMat = N_HEADS * D_HEAD * D_MODEL;  // 65536
  if (t < 3 * perMat) {
    const int mat = t / perMat;
    const int r   = t % perMat;
    const int row = r / D_MODEL;       // h*32 + n
    const int k   = r % D_MODEL;
    const int h   = row / D_HEAD;
    const int n   = row % D_HEAD;
    const float* W = (mat == 0) ? Wq : (mat == 1) ? Wk : Wv;
    Wt[t] = f2bf(W[((size_t)h * D_MODEL + k) * D_HEAD + n]);
  } else if (t < 3 * perMat + D_MODEL * D_MODEL) {
    const int r = t - 3 * perMat;
    const int n = r / D_MODEL;
    const int k = r % D_MODEL;
    Wt[t] = f2bf(Wo[(size_t)k * D_MODEL + n]);
  }
}

// ============ Kernel 1: QKV projection ====================================
// Q, K stored bf16 [B,H,S,32]; V stored TRANSPOSED bf16 [B,H,32,S].
__global__ void __launch_bounds__(32) qkv_kernel(
    const unsigned short* __restrict__ Xb,
    const unsigned short* __restrict__ Wt,   // WqT|WkT|WvT
    const float* __restrict__ bq, const float* __restrict__ bk,
    const float* __restrict__ bv,
    unsigned short* __restrict__ Q, unsigned short* __restrict__ K,
    unsigned short* __restrict__ Vt) {
  const int tiles_s = SEQ / 16;
  int blk = blockIdx.x;
  const int st = blk % tiles_s; blk /= tiles_s;
  const int h  = blk % N_HEADS; blk /= N_HEADS;
  const int b  = blk;
  const int s0 = st * 16;

  const int n  = threadIdx.x & 15;
  const int mo = ((threadIdx.x >> 4) & 1) * 8;

  // cache the 8 A-fragments of the 16x256 x-tile (reused for Q, K, V)
  v16bf a[8];
  const unsigned short* xs = Xb + (size_t)(b * SEQ + s0) * D_MODEL;
#pragma unroll
  for (int kk = 0; kk < 8; ++kk) a[kk] = frag_rows16(xs + kk * 32, D_MODEL);

  const int perMat = N_HEADS * D_HEAD * D_MODEL;
  const float* bs[3] = { bq + h * D_HEAD, bk + h * D_HEAD, bv + h * D_HEAD };

#pragma unroll
  for (int m3 = 0; m3 < 3; ++m3) {
    const unsigned short* WTh = Wt + (size_t)m3 * perMat + (size_t)h * D_HEAD * D_MODEL;
#pragma unroll
    for (int nc = 0; nc < 2; ++nc) {
      v8f acc = {};
#pragma unroll
      for (int kk = 0; kk < 8; ++kk) {
        v16bf bf = frag_rows16(WTh + (size_t)(nc * 16) * D_MODEL + kk * 32, D_MODEL);
        acc = wmma_bf16(a[kk], bf, acc);
      }
      const float bias = bs[m3][nc * 16 + n];
      if (m3 < 2) {
        unsigned short* dst = ((m3 == 0) ? Q : K) +
            ((size_t)(b * N_HEADS + h) * SEQ + s0) * D_HEAD;
#pragma unroll
        for (int r = 0; r < 8; ++r)
          dst[(size_t)(mo + r) * D_HEAD + nc * 16 + n] = f2bf(acc[r] + bias);
      } else {
        // V transposed: Vt[((b*H+h)*32 + d) * S + s]
        unsigned short* dst = Vt + ((size_t)(b * N_HEADS + h) * D_HEAD +
                                    nc * 16 + n) * SEQ + s0;
#pragma unroll
        for (int r = 0; r < 8; ++r) dst[mo + r] = f2bf(acc[r] + bias);
      }
    }
  }
}

// ============ Kernel 2: flash attention with odd-causal mask ==============
__global__ void __launch_bounds__(32) attn_kernel(
    const unsigned short* __restrict__ Q,
    const unsigned short* __restrict__ K,
    const unsigned short* __restrict__ Vt,
    unsigned short* __restrict__ Z) {
  __shared__ float          sS[16 * 32];
  __shared__ unsigned short sP[16 * 32];
  __shared__ float          sRow[16];

  const int tiles_q = SEQ / 16;
  int blk = blockIdx.x;
  const int qt = blk % tiles_q; blk /= tiles_q;
  const int h  = blk % N_HEADS; blk /= N_HEADS;
  const int b  = blk;
  const int q0 = qt * 16;

  const unsigned short* Qh  = Q  + (size_t)(b * N_HEADS + h) * SEQ * D_HEAD;
  const unsigned short* Kh  = K  + (size_t)(b * N_HEADS + h) * SEQ * D_HEAD;
  const unsigned short* Vth = Vt + (size_t)(b * N_HEADS + h) * D_HEAD * SEQ;

  const int lane = threadIdx.x;
  const int n  = lane & 15;
  const int mo = ((lane >> 4) & 1) * 8;
  const float scale = 0.17677669529663687f;  // 1/sqrt(32)

  const v16bf qa = frag_rows16(Qh + (size_t)q0 * D_HEAD, D_HEAD);

  v8f acc0 = {};
  v8f acc1 = {};
  float mrow = -1e30f, lrow = 0.0f;  // live in lanes 0..15

  const int kend = q0 + 16;  // causal bound
  for (int k0 = 0; k0 < kend; k0 += 32) {
    // prefetch next K tile (global_prefetch_b8; speculative, OOB-safe)
    __builtin_prefetch((const void*)(Kh + (size_t)(k0 + 32) * D_HEAD), 0, 0);

    // ---- scores: B-frag of K^T == A-frag of row-major K (contiguous)
    v16bf kb0 = frag_rows16(Kh + (size_t)k0 * D_HEAD, D_HEAD);
    v16bf kb1 = frag_rows16(Kh + (size_t)(k0 + 16) * D_HEAD, D_HEAD);
    v8f c0 = {};
    v8f c1 = {};
    c0 = wmma_bf16(qa, kb0, c0);
    c1 = wmma_bf16(qa, kb1, c1);
#pragma unroll
    for (int r = 0; r < 8; ++r) {
      sS[(mo + r) * 32 + n]      = c0[r];
      sS[(mo + r) * 32 + 16 + n] = c1[r];
    }
    __syncthreads();

    // ---- online softmax with mask: (kj<qi && odd kj) || kj==qi
    if (lane < 16) {
      const int row = lane;
      const int qi  = q0 + row;
      float tm = -1e30f;
      for (int j = 0; j < 32; ++j) {
        const int kj = k0 + j;
        if (((kj < qi) && (kj & 1)) || (kj == qi))
          tm = fmaxf(tm, sS[row * 32 + j] * scale);
      }
      const float mn = fmaxf(mrow, tm);
      const float c  = __expf(mrow - mn);
      float ls = 0.0f;
      for (int j = 0; j < 32; ++j) {
        const int kj = k0 + j;
        float p = 0.0f;
        if (((kj < qi) && (kj & 1)) || (kj == qi))
          p = __expf(sS[row * 32 + j] * scale - mn);
        sP[row * 32 + j] = f2bf(p);
        ls += p;
      }
      lrow = lrow * c + ls;
      mrow = mn;
      sRow[row] = c;
    }
    __syncthreads();

    // ---- rescale accumulators, then PV WMMAs (V pre-transposed => b128)
#pragma unroll
    for (int r = 0; r < 8; ++r) {
      const float cf = sRow[mo + r];
      acc0[r] *= cf;
      acc1[r] *= cf;
    }
    v16bf pf  = frag_rows16(sP, 32);                       // ds_load_b128 x2
    v16bf vb0 = frag_rows16(Vth + k0, SEQ);                // d = 0..15
    v16bf vb1 = frag_rows16(Vth + (size_t)16 * SEQ + k0, SEQ);  // d = 16..31
    acc0 = wmma_bf16(pf, vb0, acc0);
    acc1 = wmma_bf16(pf, vb1, acc1);
    __syncthreads();
  }

  // ---- finalize: divide by row sums; Z bf16 [B, S, H*32]
  if (lane < 16) sRow[lane] = 1.0f / lrow;
  __syncthreads();
  unsigned short* dst = Z + (size_t)(b * SEQ + q0) * D_MODEL + h * D_HEAD;
#pragma unroll
  for (int r = 0; r < 8; ++r) {
    const float inv = sRow[mo + r];
    dst[(size_t)(mo + r) * D_MODEL + n]      = f2bf(acc0[r] * inv);
    dst[(size_t)(mo + r) * D_MODEL + 16 + n] = f2bf(acc1[r] * inv);
  }
}

// ============ Kernel 3: output projection =================================
// out[4096,256] = Z(bf16) @ W_O + b_O ; WoT pre-transposed => b128 B-frags.
__global__ void __launch_bounds__(32) proj_kernel(
    const unsigned short* __restrict__ Z,
    const unsigned short* __restrict__ WoT,
    const float* __restrict__ bo,
    float* __restrict__ out) {
  const int tiles_n = D_MODEL / 16;
  int blk = blockIdx.x;
  const int nt = blk % tiles_n; blk /= tiles_n;
  const int r0 = blk * 16;
  const int c0 = nt * 16;

  const int n  = threadIdx.x & 15;
  const int mo = ((threadIdx.x >> 4) & 1) * 8;

  v8f acc = {};
#pragma unroll
  for (int kk = 0; kk < 8; ++kk) {
    v16bf af = frag_rows16(Z   + (size_t)r0 * D_MODEL + kk * 32, D_MODEL);
    v16bf bf = frag_rows16(WoT + (size_t)c0 * D_MODEL + kk * 32, D_MODEL);
    acc = wmma_bf16(af, bf, acc);
  }
  const float bias = bo[c0 + n];
#pragma unroll
  for (int r = 0; r < 8; ++r)
    out[(size_t)(r0 + mo + r) * D_MODEL + c0 + n] = acc[r] + bias;
}

extern "C" void kernel_launch(void* const* d_in, const int* in_sizes, int n_in,
                              void* d_out, int out_size, void* d_ws, size_t ws_size,
                              hipStream_t stream) {
  const float* x  = (const float*)d_in[0];
  const float* Wq = (const float*)d_in[1];
  const float* Wk = (const float*)d_in[2];
  const float* Wv = (const float*)d_in[3];
  const float* Wo = (const float*)d_in[4];
  const float* bq = (const float*)d_in[5];
  const float* bk = (const float*)d_in[6];
  const float* bv = (const float*)d_in[7];
  const float* bo = (const float*)d_in[8];
  float* out = (float*)d_out;

  // workspace (u16 elements): Xb | Wt(q,k,v,o) | Q | K | Vt | Z
  const size_t nX   = (size_t)BATCH * SEQ * D_MODEL;            // 1 Mi
  const size_t nWt  = (size_t)3 * N_HEADS * D_HEAD * D_MODEL +
                      (size_t)D_MODEL * D_MODEL;                // 256 Ki
  const size_t nQKV = (size_t)BATCH * N_HEADS * SEQ * D_HEAD;   // 1 Mi each
  unsigned short* Xb = (unsigned short*)d_ws;
  unsigned short* Wt = Xb + nX;
  unsigned short* Q  = Wt + nWt;
  unsigned short* K  = Q + nQKV;
  unsigned short* Vt = K + nQKV;
  unsigned short* Z  = Vt + nQKV;

  const int n8 = (int)(nX / 8);
  cvt_x_kernel<<<(n8 + 255) / 256, 256, 0, stream>>>(x, Xb, n8);
  cvt_w_kernel<<<((int)nWt + 255) / 256, 256, 0, stream>>>(Wq, Wk, Wv, Wo, Wt);

  dim3 blk(32);
  qkv_kernel<<<BATCH * N_HEADS * (SEQ / 16), blk, 0, stream>>>(
      Xb, Wt, bq, bk, bv, Q, K, Vt);
  attn_kernel<<<BATCH * N_HEADS * (SEQ / 16), blk, 0, stream>>>(Q, K, Vt, Z);
  proj_kernel<<<(BATCH * SEQ / 16) * (D_MODEL / 16), blk, 0, stream>>>(
      Z, Wt + 3 * N_HEADS * D_HEAD * D_MODEL, bo, out);
}
